// ExpertParallelMoE_32392643347047
// MI455X (gfx1250) — compile-verified, matrix operand order checked by reading, not run on hardware
//
#include <hip/hip_runtime.h>
#include <hip/hip_bf16.h>

// ---------------------------------------------------------------------------
// Expert-parallel MoE (top-2) for MI455X / gfx1250, wave32, WMMA bf16.
// T=2048 tokens, H=2048 hidden, I=5632 inter, E=8 experts, top_k=2.
// Each wave: 4 M-subtiles (64 tokens) x 16 N-cols, so every streamed weight
// fragment is reused 4x before eviction from registers.
// ---------------------------------------------------------------------------

#define T_TOK 2048
#define HID   2048
#define INTER 5632
#define NEXP  8

typedef __attribute__((ext_vector_type(16))) __bf16 v16bf;
typedef __attribute__((ext_vector_type(8)))  __bf16 v8bf;
typedef __attribute__((ext_vector_type(8)))  float  v8f;
typedef __attribute__((ext_vector_type(4)))  float  v4f;

// ---- fragment loaders matching the CDNA5 16-bit WMMA VGPR layout ----------
// lane L: half = L>>4. A-frag row m = L&15 ; B-frag weight-row n = L&15.
// elements 0..7  <- K = k0 + 8*half + (0..7)
// elements 8..15 <- K = k0 + 16 + 8*half + (0..7)

__device__ __forceinline__ v16bf frag_from_f32(const float* __restrict__ row,
                                               int k0, int half) {
  const float* p = row + k0 + half * 8;
  v4f a0 = *(const v4f*)(p);
  v4f a1 = *(const v4f*)(p + 4);
  v4f b0 = *(const v4f*)(p + 16);
  v4f b1 = *(const v4f*)(p + 20);
  v16bf r;
#pragma unroll
  for (int i = 0; i < 4; ++i) {
    r[i]      = (__bf16)a0[i];
    r[4 + i]  = (__bf16)a1[i];
    r[8 + i]  = (__bf16)b0[i];
    r[12 + i] = (__bf16)b1[i];
  }
  return r;
}

__device__ __forceinline__ v16bf frag_from_bf16(const __bf16* __restrict__ row,
                                                int k0, int half) {
  const __bf16* p = row + k0 + half * 8;
  v8bf x0 = *(const v8bf*)(p);
  v8bf x1 = *(const v8bf*)(p + 16);
  v16bf r;
#pragma unroll
  for (int i = 0; i < 8; ++i) {
    r[i]     = x0[i];
    r[8 + i] = x1[i];
  }
  return r;
}

__device__ __forceinline__ v8f wmma_bf16(v16bf a, v16bf b, v8f c) {
  return __builtin_amdgcn_wmma_f32_16x16x32_bf16(
      false, a, false, b, (short)0, c, false, false);
}

__device__ __forceinline__ float fast_silu(float g) {
  // g * 1/(1+exp(-g)) with HW rcp (v_rcp_f32) instead of IEEE divide
  return g * __builtin_amdgcn_rcpf(1.f + __expf(-g));
}

// ---------------------------------------------------------------------------
// Kernel 1: router logits + top-2 + softmax.  One 256-thread block per token.
// ---------------------------------------------------------------------------
__global__ __launch_bounds__(256) void moe_router_kernel(
    const float* __restrict__ x, const float* __restrict__ wr,
    int* __restrict__ ids, float* __restrict__ probs) {
  __shared__ float red[NEXP][256];
  const int t = blockIdx.x;
  const int tid = threadIdx.x;

  float acc[NEXP];
#pragma unroll
  for (int e = 0; e < NEXP; ++e) acc[e] = 0.f;

  for (int h = tid; h < HID; h += 256) {
    float xv = x[(long long)t * HID + h];
#pragma unroll
    for (int e = 0; e < NEXP; ++e) acc[e] += xv * wr[e * HID + h];
  }
#pragma unroll
  for (int e = 0; e < NEXP; ++e) red[e][tid] = acc[e];
  __syncthreads();
  for (int s = 128; s > 0; s >>= 1) {
    if (tid < s) {
#pragma unroll
      for (int e = 0; e < NEXP; ++e) red[e][tid] += red[e][tid + s];
    }
    __syncthreads();
  }
  if (tid == 0) {
    float v0 = -3.4e38f, v1 = -3.4e38f;
    int e0 = 0, e1 = 0;
    for (int e = 0; e < NEXP; ++e) {       // strict > : ties keep lowest index
      float v = red[e][0];
      if (v > v0)      { v1 = v0; e1 = e0; v0 = v; e0 = e; }
      else if (v > v1) { v1 = v;  e1 = e; }
    }
    float p0 = 1.f / (1.f + __expf(v1 - v0));
    ids[2 * t + 0] = e0;  probs[2 * t + 0] = p0;
    ids[2 * t + 1] = e1;  probs[2 * t + 1] = 1.f - p0;
  }
}

// ---------------------------------------------------------------------------
// Kernel 2: deterministic per-expert token lists (token order). 1 wave,
// thread e scans all tokens for expert e; thread 0 prefix-sums counts.
// ---------------------------------------------------------------------------
__global__ void moe_build_lists_kernel(
    const int* __restrict__ ids, const float* __restrict__ probs,
    int* __restrict__ cnt, int* __restrict__ off,
    int* __restrict__ slot_tok, float* __restrict__ slot_p,
    int* __restrict__ slot_k) {
  const int e = threadIdx.x;
  if (e < NEXP) {
    int c = 0;
    for (int t = 0; t < T_TOK; ++t) {
#pragma unroll
      for (int k = 0; k < 2; ++k) {
        if (ids[2 * t + k] == e) {
          slot_tok[e * T_TOK + c] = t;
          slot_p[e * T_TOK + c]   = probs[2 * t + k];
          slot_k[e * T_TOK + c]   = k;
          ++c;
        }
      }
    }
    cnt[e] = c;
  }
  __syncthreads();
  if (e == 0) {
    int s = 0;
    for (int i = 0; i < NEXP; ++i) { off[i] = s; s += cnt[i]; }
  }
}

// ---------------------------------------------------------------------------
// Kernel 3: gate/up GEMMs + SiLU -> bf16 act in workspace (compact slots).
// Block = 8 waves; wave w owns a 16-wide inter tile; block covers 128 cols
// x 64 tokens (4 M-subtiles per wave reuse each B fragment 4x).
// grid = (INTER/128, T_TOK/64, NEXP)
// ---------------------------------------------------------------------------
__global__ __launch_bounds__(256) void moe_gateup_kernel(
    const float* __restrict__ x, const float* __restrict__ wg,
    const float* __restrict__ wu, const int* __restrict__ cnt,
    const int* __restrict__ off, const int* __restrict__ slot_tok,
    __bf16* __restrict__ act) {
  const int e   = blockIdx.z;
  const int mt  = blockIdx.y;      // 64-token tile
  const int n_e = cnt[e];
  if (mt * 64 >= n_e) return;

  const int wave = threadIdx.x >> 5;
  const int lane = threadIdx.x & 31;
  const int half = lane >> 4;
  const int l16  = lane & 15;
  const int nbase = blockIdx.x * 128 + wave * 16;

  const float* arow[4];
#pragma unroll
  for (int m = 0; m < 4; ++m) {
    int pos = mt * 64 + m * 16 + l16;
    if (pos > n_e - 1) pos = n_e - 1;            // clamp padded rows
    arow[m] = x + (long long)slot_tok[e * T_TOK + pos] * HID;
  }
  const float* grow = wg + ((long long)e * INTER + nbase + l16) * HID;
  const float* urow = wu + ((long long)e * INTER + nbase + l16) * HID;

  v8f cg[4] = {}, cu[4] = {};
  for (int k0 = 0; k0 < HID; k0 += 32) {
    v16bf bg = frag_from_f32(grow, k0, half);
    v16bf bu = frag_from_f32(urow, k0, half);
#pragma unroll
    for (int m = 0; m < 4; ++m) {
      v16bf a = frag_from_f32(arow[m], k0, half);
      cg[m] = wmma_bf16(a, bg, cg[m]);
      cu[m] = wmma_bf16(a, bu, cu[m]);
    }
  }

  const int slot_base = off[e];
  const int ncol = nbase + l16;
#pragma unroll
  for (int m = 0; m < 4; ++m) {
#pragma unroll
    for (int r = 0; r < 8; ++r) {
      const int prow = mt * 64 + m * 16 + r + half * 8;
      if (prow < n_e) {
        act[(long long)(slot_base + prow) * INTER + ncol] =
            (__bf16)(fast_silu(cg[m][r]) * cu[m][r]);
      }
    }
  }
}

// ---------------------------------------------------------------------------
// Kernel 4: down GEMM + combine-weight scale. Writes per-k output buffers
// (each (token,k) pair occurs exactly once -> no atomics, deterministic).
// grid = (HID/128, T_TOK/64, NEXP)
// ---------------------------------------------------------------------------
__global__ __launch_bounds__(256) void moe_down_kernel(
    const __bf16* __restrict__ act, const float* __restrict__ wd,
    const int* __restrict__ cnt, const int* __restrict__ off,
    const int* __restrict__ slot_tok, const float* __restrict__ slot_p,
    const int* __restrict__ slot_k,
    float* __restrict__ out0, float* __restrict__ out1) {
  const int e   = blockIdx.z;
  const int mt  = blockIdx.y;
  const int n_e = cnt[e];
  if (mt * 64 >= n_e) return;

  const int wave = threadIdx.x >> 5;
  const int lane = threadIdx.x & 31;
  const int half = lane >> 4;
  const int l16  = lane & 15;
  const int hbase = blockIdx.x * 128 + wave * 16;
  const int slot_base = off[e];

  const __bf16* arow[4];
#pragma unroll
  for (int m = 0; m < 4; ++m) {
    int pos = mt * 64 + m * 16 + l16;
    if (pos > n_e - 1) pos = n_e - 1;
    arow[m] = act + (long long)(slot_base + pos) * INTER;
  }
  const float* drow = wd + ((long long)e * HID + hbase + l16) * INTER;

  v8f c[4] = {};
  for (int k0 = 0; k0 < INTER; k0 += 32) {
    v16bf b = frag_from_f32(drow, k0, half);
#pragma unroll
    for (int m = 0; m < 4; ++m) {
      v16bf a = frag_from_bf16(arow[m], k0, half);
      c[m] = wmma_bf16(a, b, c[m]);
    }
  }

  const int hcol = hbase + l16;
#pragma unroll
  for (int m = 0; m < 4; ++m) {
#pragma unroll
    for (int r = 0; r < 8; ++r) {
      const int prow = mt * 64 + m * 16 + r + half * 8;
      if (prow < n_e) {
        const int   tok = slot_tok[e * T_TOK + prow];
        const float p   = slot_p[e * T_TOK + prow];
        float* dst = slot_k[e * T_TOK + prow] ? out1 : out0;
        dst[(long long)tok * HID + hcol] = p * c[m][r];
      }
    }
  }
}

// ---------------------------------------------------------------------------
// Kernel 5: out = out0 + out1
// ---------------------------------------------------------------------------
__global__ void moe_combine_kernel(const float* __restrict__ o0,
                                   const float* __restrict__ o1,
                                   float* __restrict__ out) {
  const long long n = (long long)T_TOK * HID;
  for (long long i = blockIdx.x * (long long)blockDim.x + threadIdx.x; i < n;
       i += (long long)gridDim.x * blockDim.x)
    out[i] = o0[i] + o1[i];
}

// ---------------------------------------------------------------------------
extern "C" void kernel_launch(void* const* d_in, const int* in_sizes, int n_in,
                              void* d_out, int out_size, void* d_ws,
                              size_t ws_size, hipStream_t stream) {
  const float* x  = (const float*)d_in[0];
  const float* wr = (const float*)d_in[1];
  const float* wg = (const float*)d_in[2];
  const float* wu = (const float*)d_in[3];
  const float* wd = (const float*)d_in[4];
  // d_in[5] = top_k (assumed 2, matching the reference setup)
  float* out = (float*)d_out;

  char* ws = (char*)d_ws;
  // workspace layout (bytes)
  int*    ids      = (int*)   (ws + 0);                       // T*2 ints
  float*  probs    = (float*) (ws + 16384);                   // T*2 floats
  int*    cnt      = (int*)   (ws + 32768);                   // 8
  int*    off      = (int*)   (ws + 32832);                   // 8
  int*    slot_tok = (int*)   (ws + 32896);                   // E*T
  float*  slot_p   = (float*) (ws + 98432);                   // E*T
  int*    slot_k   = (int*)   (ws + 163968);                  // E*T
  __bf16* act      = (__bf16*)(ws + 229504);                  // 4096*INTER bf16
  float*  out0     = (float*) (ws + 229504 + (size_t)4096 * INTER * 2);
  float*  out1     = out0 + (size_t)T_TOK * HID;

  moe_router_kernel<<<T_TOK, 256, 0, stream>>>(x, wr, ids, probs);
  moe_build_lists_kernel<<<1, 32, 0, stream>>>(ids, probs, cnt, off, slot_tok,
                                               slot_p, slot_k);
  moe_gateup_kernel<<<dim3(INTER / 128, T_TOK / 64, NEXP), 256, 0, stream>>>(
      x, wg, wu, cnt, off, slot_tok, act);
  moe_down_kernel<<<dim3(HID / 128, T_TOK / 64, NEXP), 256, 0, stream>>>(
      act, wd, cnt, off, slot_tok, slot_p, slot_k, out0, out1);
  moe_combine_kernel<<<1024, 256, 0, stream>>>(out0, out1, out);
}
